// ChebyshevSameInfluenceProcess_3410204033611
// MI455X (gfx1250) — matrix operand order repaired and checked by reading, not compiled, not run on hardware
//
#include <hip/hip_runtime.h>
#include <math.h>

#define B_SZ   32
#define SEQLEN 256
#define HID    64
#define TEND   50.0f
#define NCHEB  1000
#define PI_F   3.14159265358979323846f

typedef __attribute__((ext_vector_type(16))) _Float16 v16h;
typedef __attribute__((ext_vector_type(8)))  _Float16 v8h;
typedef __attribute__((ext_vector_type(8)))  float    v8f;

#define LDS_WAIT() asm volatile("s_wait_dscnt 0x0" ::: "memory")

// Branch-free tanh: sign(x)*(1-e)/(1+e), e = exp(-2|x|).
// Saturates to +-1 naturally for large |x| (e -> 0); no EXEC divergence.
__device__ __forceinline__ float tanh_fast(float x) {
  float e = __expf(-2.0f * fabsf(x));
  float t = (1.0f - e) * __builtin_amdgcn_rcpf(1.0f + e);
  return copysignf(t, x);
}

// Branch-free stable softplus: max(x,0) + log(1 + exp(-|x|)).
__device__ __forceinline__ float softplus_fast(float x) {
  return fmaxf(x, 0.f) + __logf(1.0f + __expf(-fabsf(x)));
}

// B-fragments for both 64x64 layers, held in VGPRs for the whole block:
// w[layer][nt][kc] is the 32x16 chunk (K = kc*32.., N = nt*16..).
struct BF { v16h w[2][4][2]; };

// ---------------------------------------------------------------------------
// Load all W2/W3 B-fragments straight from global (L2-resident) into regs.
// ISA B layout (32x16 f16): lane l holds N = n0+(l&15); halfs j<8 = B[kb+j][N],
// halfs j>=8 = B[kb+8+j][N], with kb = k0 + ((l&16)?8:0).
// ---------------------------------------------------------------------------
__device__ __forceinline__ void load_bfrags(int lane, const float* W2,
                                            const float* W3, BF& bf) {
  int n  = lane & 15;
  int kh = (lane & 16) ? 8 : 0;
#pragma unroll
  for (int nt = 0; nt < 4; ++nt)
#pragma unroll
    for (int kc = 0; kc < 2; ++kc) {
      int kb = kc * 32 + kh;
#pragma unroll
      for (int j = 0; j < 16; ++j) {
        int kk = kb + j + ((j >= 8) ? 8 : 0);
        bf.w[0][nt][kc][j] = (_Float16)W2[kk * HID + nt * 16 + n];
        bf.w[1][nt][kc][j] = (_Float16)W3[kk * HID + nt * 16 + n];
      }
    }
}

// ---------------------------------------------------------------------------
// A-fragment loader: 16x32 f16 A chunk from a row-major 16x64 LDS tile.
// Lane l<16: row l, K = k0..k0+7 (halfs 0-7), K = k0+16..k0+23 (halfs 8-15);
// lanes 16-31: same rows, +8 K offset.  Two ds_load_b128 per fragment.
// ---------------------------------------------------------------------------
__device__ __forceinline__ v16h load_a_frag(const _Float16* h, int k0, int lane) {
  int row = lane & 15;
  int kb  = k0 + ((lane & 16) ? 8 : 0);
  const _Float16* p = h + row * HID + kb;
  v8h lo = *(const v8h*)(p);
  v8h hi = *(const v8h*)(p + 16);
  v16h a;
#pragma unroll
  for (int j = 0; j < 8; ++j) { a[j] = lo[j]; a[j + 8] = hi[j]; }
  return a;
}

// Stage the small per-neuron vectors into LDS (single wave, LDS in-order).
__device__ __forceinline__ void stage_vectors(int lane, const float* W1,
                                              const float* b1, const float* b2,
                                              const float* b3, const float* W4,
                                              float* W1s, float* b1s, float* b2s,
                                              float* b3s, float* W4s) {
  for (int idx = lane; idx < HID; idx += 32) {
    W1s[idx] = W1[idx]; b1s[idx] = b1[idx];
    b2s[idx] = b2[idx]; b3s[idx] = b3[idx];
    W4s[idx] = W4[idx];
  }
  LDS_WAIT();
}

// ---------------------------------------------------------------------------
// 4-layer MLP for a tile of 16 scalar inputs, one wave.
// x: per-lane input for point p = lane&15 (halves mirror each other).
// Layer 1 writes its tanh outputs DIRECTLY into the A-fragment register
// layout (lane l owns exactly the (row, K) elements of its own fragment),
// so only the layer2->layer3 and layer3->layer4 shuffles touch LDS.
// Returns softplus(MLP(x)) on every lane.
// ---------------------------------------------------------------------------
__device__ __forceinline__ float mlp_tile(int lane, float x, const float* W1s,
                                          const float* b1s, const float* b2s,
                                          const float* b3s, const float* W4s,
                                          float b4, const BF& bf,
                                          _Float16 (*hT)[HID]) {
  int p   = lane & 15;
  int jo  = (lane & 16) ? 8 : 0;
  int n16 = lane & 15;
  int mb  = (lane & 16) ? 8 : 0;

  // ---- layer 1: x -> 64, tanh, straight into A fragments ----
  v16h a0, a1;
#pragma unroll
  for (int g = 0; g < 2; ++g)
#pragma unroll
    for (int u = 0; u < 8; ++u) {
      int j0 = jo + g * 16 + u;       // K index for a0 half 'g'
      int j1 = j0 + 32;               // K index for a1 half 'g'
      a0[g * 8 + u] = (_Float16)tanh_fast(fmaf(x, W1s[j0], b1s[j0]));
      a1[g * 8 + u] = (_Float16)tanh_fast(fmaf(x, W1s[j1], b1s[j1]));
    }

  // ---- layer 2: (16x64) @ W2 -> tanh -> hT ----
#pragma unroll
  for (int nt = 0; nt < 4; ++nt) {
    float bn = b2s[nt * 16 + n16];
    v8f acc;
#pragma unroll
    for (int r = 0; r < 8; ++r) acc[r] = bn;
    acc = __builtin_amdgcn_wmma_f32_16x16x32_f16(false, a0, false, bf.w[0][nt][0],
                                                 (short)0, acc, false, false);
    acc = __builtin_amdgcn_wmma_f32_16x16x32_f16(false, a1, false, bf.w[0][nt][1],
                                                 (short)0, acc, false, false);
    int n = nt * 16 + n16;
#pragma unroll
    for (int r = 0; r < 8; ++r) hT[mb + r][n] = (_Float16)tanh_fast(acc[r]);
  }
  LDS_WAIT();

  // ---- layer 3: reload A frags, (16x64) @ W3 -> tanh -> hT ----
  a0 = load_a_frag(&hT[0][0], 0, lane);
  a1 = load_a_frag(&hT[0][0], 32, lane);
#pragma unroll
  for (int nt = 0; nt < 4; ++nt) {
    float bn = b3s[nt * 16 + n16];
    v8f acc;
#pragma unroll
    for (int r = 0; r < 8; ++r) acc[r] = bn;
    acc = __builtin_amdgcn_wmma_f32_16x16x32_f16(false, a0, false, bf.w[1][nt][0],
                                                 (short)0, acc, false, false);
    acc = __builtin_amdgcn_wmma_f32_16x16x32_f16(false, a1, false, bf.w[1][nt][1],
                                                 (short)0, acc, false, false);
    int n = nt * 16 + n16;
#pragma unroll
    for (int r = 0; r < 8; ++r) hT[mb + r][n] = (_Float16)tanh_fast(acc[r]);
  }
  LDS_WAIT();

  // ---- layer 4: 64 -> 1, split across half-waves, softplus ----
  float s = 0.f;
  int j0 = (lane & 16) ? 32 : 0;
#pragma unroll
  for (int jj = 0; jj < 32; ++jj)
    s = fmaf((float)hT[p][j0 + jj], W4s[j0 + jj], s);
  s += __shfl_xor(s, 16, 32);   // combine the two half-sums
  return softplus_fast(s + b4);
}

// ===========================================================================
// Kernel 1: MLP at the 1000 Chebyshev nodes (WMMA path), fv[k] = f(x_k)
// ===========================================================================
__global__ __launch_bounds__(32) void cheb_nodes_kernel(
    const float* __restrict__ W1, const float* __restrict__ b1,
    const float* __restrict__ W2, const float* __restrict__ b2,
    const float* __restrict__ W3, const float* __restrict__ b3,
    const float* __restrict__ W4, const float* __restrict__ b4p,
    float* __restrict__ fv) {
  __shared__ __align__(16) _Float16 hT[16][HID];
  __shared__ float W1s[HID], b1s[HID], b2s[HID], b3s[HID], W4s[HID];
  int lane = threadIdx.x;
  stage_vectors(lane, W1, b1, b2, b3, W4, W1s, b1s, b2s, b3s, W4s);
  BF bf;
  load_bfrags(lane, W2, W3, bf);
  int k = blockIdx.x * 16 + (lane & 15);
  float th = PI_F * ((float)k + 0.5f) / (float)NCHEB;
  float x  = (k < NCHEB) ? 0.5f * TEND * (cosf(th) + 1.0f) : 0.f;
  float f = mlp_tile(lane, x, W1s, b1s, b2s, b3s, W4s, b4p[0], bf, hT);
  if (lane < 16 && k < NCHEB) fv[k] = f;
}

// ===========================================================================
// Kernel 2: Chebyshev transform c[j] = 2/n * sum_k cos(j*theta_k)*fv[k]
// ===========================================================================
__global__ __launch_bounds__(256) void cheb_c_kernel(
    const float* __restrict__ fv, float* __restrict__ cvec) {
  __shared__ float fvs[NCHEB];
  for (int idx = threadIdx.x; idx < NCHEB; idx += 256) fvs[idx] = fv[idx];
  __syncthreads();
  int j = blockIdx.x * 256 + threadIdx.x;
  if (j < NCHEB) {
    float jf = (float)j, s = 0.f;
    for (int k = 0; k < NCHEB; ++k) {
      float th = PI_F * ((float)k + 0.5f) / (float)NCHEB;
      s = fmaf(cosf(jf * th), fvs[k], s);
    }
    cvec[j] = (2.0f / (float)NCHEB) * s;
  }
}

// ===========================================================================
// Kernel 3: analytic integral coefficients (chint): cint[0..999]
// ===========================================================================
__global__ __launch_bounds__(256) void cheb_int_kernel(
    const float* __restrict__ cvec, float* __restrict__ cint) {
  __shared__ float red[256];
  const float con = 0.25f * TEND;
  float part = 0.f;
  for (int m = threadIdx.x; m < NCHEB - 2; m += 256) {   // m = 0..997
    float tl = con * (cvec[m] - cvec[m + 2]) / (float)(m + 1);
    cint[m + 1] = tl;
    part += (m & 1) ? -tl : tl;
  }
  if (threadIdx.x == 0) {
    float tl = con * cvec[NCHEB - 2] / (float)(NCHEB - 1);
    cint[NCHEB - 1] = tl;                                 // m = 998 (even)
    part += tl;
  }
  red[threadIdx.x] = part;
  __syncthreads();
  for (int s = 128; s; s >>= 1) {
    if (threadIdx.x < s) red[threadIdx.x] += red[threadIdx.x + s];
    __syncthreads();
  }
  if (threadIdx.x == 0) cint[0] = 2.0f * red[0];
}

// ===========================================================================
// Kernel 4 (the big one): one wave per (b, i), i = 1..255.
// lambda = bg + sum_{c<i} f(t_i - t_c), MLP via WMMA on 16-point tiles.
// Writes loglam[b*255 + (i-1)] = mask ? log(lambda) : 0.
// ===========================================================================
__global__ __launch_bounds__(32) void pair_nll_kernel(
    const float* __restrict__ seq_pads, const int* __restrict__ seq_lens,
    const float* __restrict__ W1, const float* __restrict__ b1,
    const float* __restrict__ W2, const float* __restrict__ b2,
    const float* __restrict__ W3, const float* __restrict__ b3,
    const float* __restrict__ W4, const float* __restrict__ b4p,
    const float* __restrict__ bgp, float* __restrict__ loglam) {
  __shared__ __align__(16) _Float16 hT[16][HID];
  __shared__ float W1s[HID], b1s[HID], b2s[HID], b3s[HID], W4s[HID];
  __shared__ float tb[SEQLEN];

  int lane = threadIdx.x;
  int i = blockIdx.x + 1;   // 1..255
  int b = blockIdx.y;

  for (int idx = lane; idx < SEQLEN; idx += 32) tb[idx] = seq_pads[b * SEQLEN + idx];
  stage_vectors(lane, W1, b1, b2, b3, W4, W1s, b1s, b2s, b3s, W4s);
  BF bf;
  load_bfrags(lane, W2, W3, bf);   // weights stay in VGPRs for all tiles

  float b4 = b4p[0];
  float ti = tb[i];
  float acc = 0.f;
  for (int c0 = 0; c0 < i; c0 += 16) {
    int c = c0 + (lane & 15);             // c <= 255 always
    float x = ti - tb[c];
    float f = mlp_tile(lane, x, W1s, b1s, b2s, b3s, W4s, b4, bf, hT);
    if (lane < 16 && c < i) acc += f;     // lanes 16-31 contribute 0 (dupes)
  }
#pragma unroll
  for (int off = 16; off >= 1; off >>= 1) acc += __shfl_xor(acc, off, 32);
  if (lane == 0) {
    float lam = bgp[0] + acc;
    loglam[b * (SEQLEN - 1) + (i - 1)] = (i < seq_lens[b]) ? logf(lam) : 0.f;
  }
}

// ===========================================================================
// Kernel 5: Clenshaw evaluation of the integral series at (T-t) per point.
// cos(j*arccos(y)) == T_j(y), so Clenshaw matches the reference formula.
// ===========================================================================
__global__ __launch_bounds__(256) void chebev_kernel(
    const float* __restrict__ seq_pads, const int* __restrict__ seq_lens,
    const float* __restrict__ cint, float* __restrict__ chebres) {
  __shared__ float cs[NCHEB];
  int b = blockIdx.x, i = threadIdx.x;
  for (int idx = i; idx < NCHEB; idx += 256) cs[idx] = cint[idx];
  __syncthreads();
  float t = seq_pads[b * SEQLEN + i];
  float x = (i < seq_lens[b]) ? (TEND - t) : 0.f;
  float y = fminf(fmaxf(2.f * x / TEND - 1.f, -1.f), 1.f);
  float y2 = 2.f * y, d1 = 0.f, d2 = 0.f;
  for (int j = NCHEB - 1; j >= 1; --j) {
    float d0 = fmaf(y2, d1, cs[j] - d2);
    d2 = d1; d1 = d0;
  }
  chebres[b * SEQLEN + i] = fmaf(y, d1, 0.5f * cs[0] - d2);
}

// ===========================================================================
// Kernel 6: deterministic fixed-order reduction + final NLL
// ===========================================================================
__global__ __launch_bounds__(256) void finalize_kernel(
    const int* __restrict__ seq_lens, const float* __restrict__ bgp,
    const float* __restrict__ loglam, const float* __restrict__ chebres,
    float* __restrict__ out) {
  __shared__ float red[256];
  int tid = threadIdx.x;
  float s = 0.f;
  for (int idx = tid; idx < B_SZ * (SEQLEN - 1); idx += 256) s += loglam[idx];
  for (int idx = tid; idx < B_SZ * SEQLEN; idx += 256) s -= chebres[idx];
  red[tid] = s;
  __syncthreads();
  for (int k = 128; k; k >>= 1) {
    if (tid < k) red[tid] += red[tid + k];
    __syncthreads();
  }
  if (tid == 0) {
    float bg = bgp[0];
    float tot = red[0];
    for (int b = 0; b < B_SZ; ++b)             // i = 0 rows: lambda == bg
      if (seq_lens[b] > 0) tot += logf(bg);
    tot -= (float)B_SZ * TEND * bg;            // + T*bg per batch in integral
    out[0] = -tot / (float)B_SZ;
  }
}

// ===========================================================================
extern "C" void kernel_launch(void* const* d_in, const int* in_sizes, int n_in,
                              void* d_out, int out_size, void* d_ws, size_t ws_size,
                              hipStream_t stream) {
  (void)in_sizes; (void)n_in; (void)out_size; (void)ws_size;
  const float* seq_pads = (const float*)d_in[0];
  const int*   seq_lens = (const int*)d_in[1];
  const float* W1 = (const float*)d_in[2];
  const float* b1 = (const float*)d_in[3];
  const float* W2 = (const float*)d_in[4];
  const float* b2 = (const float*)d_in[5];
  const float* W3 = (const float*)d_in[6];
  const float* b3 = (const float*)d_in[7];
  const float* W4 = (const float*)d_in[8];
  const float* b4 = (const float*)d_in[9];
  const float* bg = (const float*)d_in[10];
  float* out = (float*)d_out;

  float* ws      = (float*)d_ws;
  float* fv      = ws;                      // 1000 (pad 1024)
  float* cvec    = ws + 1024;               // 1000 (pad 1024)
  float* cint    = ws + 2048;               // 1000 (pad 1024)
  float* loglam  = ws + 3072;               // 32*255 = 8160
  float* chebres = ws + 3072 + 8192;        // 32*256 = 8192

  cheb_nodes_kernel<<<dim3((NCHEB + 15) / 16), dim3(32), 0, stream>>>(
      W1, b1, W2, b2, W3, b3, W4, b4, fv);
  cheb_c_kernel<<<dim3(4), dim3(256), 0, stream>>>(fv, cvec);
  cheb_int_kernel<<<dim3(1), dim3(256), 0, stream>>>(cvec, cint);
  pair_nll_kernel<<<dim3(SEQLEN - 1, B_SZ), dim3(32), 0, stream>>>(
      seq_pads, seq_lens, W1, b1, W2, b2, W3, b3, W4, b4, bg, loglam);
  chebev_kernel<<<dim3(B_SZ), dim3(256), 0, stream>>>(seq_pads, seq_lens, cint, chebres);
  finalize_kernel<<<dim3(1), dim3(256), 0, stream>>>(seq_lens, bg, loglam, chebres, out);
}
